// GRUCell_52201032516001
// MI455X (gfx1250) — matrix operand order repaired and checked by reading, not compiled, not run on hardware
//
#include <hip/hip_runtime.h>
#include <hip/hip_bf16.h>

#define B_  64
#define T_  1024
#define I_  512
#define H_  512
#define G3  (3 * H_)     // 1536 gate rows
#define KC  (I_ + H_)    // 1024 concatenated K

typedef __attribute__((ext_vector_type(16))) __bf16 v16bf;
typedef __attribute__((ext_vector_type(8)))  float  v8f;

union V16 {
    v16bf bf;
    __bf16 h[16];
    unsigned short us[16];
    uint4 q[2];
};

static __device__ __forceinline__ unsigned short f2bf(float f) {
    unsigned u = __float_as_uint(f);
    u += 0x7FFFu + ((u >> 16) & 1u);   // round-to-nearest-even
    return (unsigned short)(u >> 16);
}
static __device__ __forceinline__ float sigmoidf_(float x) {
    return 1.0f / (1.0f + __expf(-x));
}
static __device__ __forceinline__ float tanhf_(float x) {
    x = fminf(fmaxf(x, -15.0f), 15.0f);
    float e = __expf(2.0f * x);
    return (e - 1.0f) / (e + 1.0f);
}

// ---- workspace layout (bytes) ----
#define WC_OFF    ((size_t)0)               // bf16 [1536][1024] packed [W_i | W_h]
#define WC_BYTES  ((size_t)G3 * KC * 2)     // 3,145,728
#define BC_OFF    (WC_OFF + WC_BYTES)       // fp32 [1536] combined bias
#define BC_BYTES  ((size_t)G3 * 4)
#define HB0_OFF   (BC_OFF + BC_BYTES)       // bf16 h ping
#define HB_BYTES  ((size_t)B_ * H_ * 2)
#define HB1_OFF   (HB0_OFF + HB_BYTES)      // bf16 h pong
#define CNT_OFF   (((HB1_OFF + HB_BYTES) + 63) & ~(size_t)63)  // barrier counter
#define XBF_OFF   (((CNT_OFF + 64) + 255) & ~(size_t)255)      // optional bf16 x
#define XBF_BYTES ((size_t)B_ * T_ * I_ * 2)                   // 64 MB
#define WS_NEED_XBF (XBF_OFF + XBF_BYTES)

#define NBLOCKS 16
#define NTHREADS 256

// Pack [W_i | W_h] -> bf16 row-major [g][1024]; combined bias = b_i + b_h.
__global__ void gru_pack_kernel(const float* __restrict__ Wi,
                                const float* __restrict__ Wh,
                                const float* __restrict__ bi,
                                const float* __restrict__ bh,
                                unsigned short* __restrict__ Wc,
                                float* __restrict__ bc) {
    int idx = blockIdx.x * blockDim.x + threadIdx.x;
    if (idx < G3 * KC) {
        int g = idx / KC, k = idx % KC;
        float v = (k < I_) ? Wi[(size_t)g * I_ + k] : Wh[(size_t)g * H_ + (k - I_)];
        Wc[idx] = f2bf(v);
    }
    if (idx < G3) bc[idx] = bi[idx] + bh[idx];
}

// Parallel pre-pass: x fp32 -> bf16 (8 elements per thread), bandwidth bound.
__global__ void gru_xpack_kernel(const float* __restrict__ x,
                                 unsigned short* __restrict__ xbf) {
    size_t i = ((size_t)blockIdx.x * blockDim.x + threadIdx.x) * 8;
    if (i < (size_t)B_ * T_ * I_) {
        float4 f0 = *(const float4*)(x + i);
        float4 f1 = *(const float4*)(x + i + 4);
        unsigned short o[8];
        o[0] = f2bf(f0.x); o[1] = f2bf(f0.y); o[2] = f2bf(f0.z); o[3] = f2bf(f0.w);
        o[4] = f2bf(f1.x); o[5] = f2bf(f1.y); o[6] = f2bf(f1.z); o[7] = f2bf(f1.w);
        *(uint4*)(xbf + i) = *(const uint4*)o;
    }
}

// Initialize bf16 h ping buffer and reset barrier counter.
__global__ void gru_init_kernel(const float* __restrict__ h0,
                                unsigned short* __restrict__ hb0,
                                unsigned* __restrict__ cnt) {
    int idx = blockIdx.x * blockDim.x + threadIdx.x;
    if (idx < B_ * H_) hb0[idx] = f2bf(h0[idx]);
    if (idx == 0) *cnt = 0u;
}

// Persistent fused GRU scan: one 16x16 h-tile per wave, grid barrier per step.
// fp32 carry state (z*h_prev path) lives in registers; only bf16 h round-trips.
__launch_bounds__(NTHREADS)
__global__ void gru_scan_kernel(const float* __restrict__ x,           // [B,T,I] fp32
                                const unsigned short* __restrict__ xbf, // bf16 mirror (optional)
                                int use_xbf,
                                const unsigned short* __restrict__ Wc,  // [1536][1024] bf16
                                const float* __restrict__ bc,           // [1536]
                                const float* __restrict__ h0,           // [B,H] fp32
                                unsigned short* __restrict__ hb0,
                                unsigned short* __restrict__ hb1,
                                float* __restrict__ y,                  // [B,T,H]
                                float* __restrict__ hlast,              // [B,H]
                                unsigned* __restrict__ cnt) {
    const int lane = threadIdx.x & 31;
    const int wave = (blockIdx.x * blockDim.x + threadIdx.x) >> 5;  // 0..127
    const int rt   = wave >> 5;   // 0..3   : b-row tile
    const int ct   = wave & 31;   // 0..31  : h-col tile
    const int hf   = lane >> 4;   // lane half (0/1)
    const int ln   = lane & 15;

    // Gate rows (B-operand rows) owned by this lane: z, r, c share column n=ln.
    const int gz = ct * 16 + ln;
    const float bzv = bc[gz];
    const float brv = bc[gz + H_];
    const float bcv = bc[gz + 2 * H_];
    const unsigned short* wz = Wc + (size_t)gz * KC;
    const unsigned short* wr = Wc + (size_t)(gz + H_) * KC;
    const unsigned short* wq = Wc + (size_t)(gz + 2 * H_) * KC;

    // A-operand row (matrix row m = ln for this lane), absolute batch row:
    const int arow = rt * 16 + ln;

    // D-matrix element coords for this lane: m = v + 8*hf, n = ln.
    const int bbase = rt * 16 + hf * 8;
    const int j = ct * 16 + ln;

    // fp32 carry state in registers (this wave owns tile (b,j) for all t).
    float hreg[8];
    #pragma unroll
    for (int v = 0; v < 8; ++v) hreg[v] = h0[(size_t)(bbase + v) * H_ + j];

    for (int t = 0; t < T_; ++t) {
        const unsigned short* hrd = (t & 1) ? hb1 : hb0;   // t=0 reads ping (init)
        unsigned short*       hwr = (t & 1) ? hb0 : hb1;

        v8f az = {bzv, bzv, bzv, bzv, bzv, bzv, bzv, bzv};
        v8f ar = {brv, brv, brv, brv, brv, brv, brv, brv};
        v8f ac = {bcv, bcv, bcv, bcv, bcv, bcv, bcv, bcv};

        const unsigned short* hrow = hrd + (size_t)arow * H_;

        // ---- K-blocks 0..15: x_t contribution ----
        if (use_xbf) {
            const unsigned short* xrow = xbf + ((size_t)arow * T_ + t) * I_;
            #pragma unroll 4
            for (int kb = 0; kb < 16; ++kb) {
                const int k0 = kb * 32 + hf * 16; // 16 contiguous bf16 per lane
                V16 a;
                a.q[0] = *(const uint4*)(xrow + k0);
                a.q[1] = *(const uint4*)(xrow + k0 + 8);

                const int kw = kb * 32 + hf * 16; // B: 16 contiguous bf16 per lane
                V16 vz, vr, vq;
                vz.q[0] = *(const uint4*)(wz + kw); vz.q[1] = *(const uint4*)(wz + kw + 8);
                vr.q[0] = *(const uint4*)(wr + kw); vr.q[1] = *(const uint4*)(wr + kw + 8);
                vq.q[0] = *(const uint4*)(wq + kw); vq.q[1] = *(const uint4*)(wq + kw + 8);

                az = __builtin_amdgcn_wmma_f32_16x16x32_bf16(false, a.bf, false, vz.bf, (short)0, az, false, false);
                ar = __builtin_amdgcn_wmma_f32_16x16x32_bf16(false, a.bf, false, vr.bf, (short)0, ar, false, false);
                ac = __builtin_amdgcn_wmma_f32_16x16x32_bf16(false, a.bf, false, vq.bf, (short)0, ac, false, false);
            }
        } else {
            const float* xrow = x + ((size_t)arow * T_ + t) * I_;
            #pragma unroll 4
            for (int kb = 0; kb < 16; ++kb) {
                const int k0 = kb * 32 + hf * 16;
                V16 a;
                float4 f0 = *(const float4*)(xrow + k0);
                float4 f1 = *(const float4*)(xrow + k0 + 4);
                float4 f2 = *(const float4*)(xrow + k0 + 8);
                float4 f3 = *(const float4*)(xrow + k0 + 12);
                a.h[0]  = (__bf16)f0.x; a.h[1]  = (__bf16)f0.y;
                a.h[2]  = (__bf16)f0.z; a.h[3]  = (__bf16)f0.w;
                a.h[4]  = (__bf16)f1.x; a.h[5]  = (__bf16)f1.y;
                a.h[6]  = (__bf16)f1.z; a.h[7]  = (__bf16)f1.w;
                a.h[8]  = (__bf16)f2.x; a.h[9]  = (__bf16)f2.y;
                a.h[10] = (__bf16)f2.z; a.h[11] = (__bf16)f2.w;
                a.h[12] = (__bf16)f3.x; a.h[13] = (__bf16)f3.y;
                a.h[14] = (__bf16)f3.z; a.h[15] = (__bf16)f3.w;

                const int kw = kb * 32 + hf * 16;
                V16 vz, vr, vq;
                vz.q[0] = *(const uint4*)(wz + kw); vz.q[1] = *(const uint4*)(wz + kw + 8);
                vr.q[0] = *(const uint4*)(wr + kw); vr.q[1] = *(const uint4*)(wr + kw + 8);
                vq.q[0] = *(const uint4*)(wq + kw); vq.q[1] = *(const uint4*)(wq + kw + 8);

                az = __builtin_amdgcn_wmma_f32_16x16x32_bf16(false, a.bf, false, vz.bf, (short)0, az, false, false);
                ar = __builtin_amdgcn_wmma_f32_16x16x32_bf16(false, a.bf, false, vr.bf, (short)0, ar, false, false);
                ac = __builtin_amdgcn_wmma_f32_16x16x32_bf16(false, a.bf, false, vq.bf, (short)0, ac, false, false);
            }
        }

        // ---- K-blocks 16..31: h_prev contribution (bf16 ping-pong buffer) ----
        #pragma unroll 4
        for (int kb = 0; kb < 16; ++kb) {
            const int kh = kb * 32 + hf * 16;
            V16 a;
            a.q[0] = *(const uint4*)(hrow + kh);
            a.q[1] = *(const uint4*)(hrow + kh + 8);

            const int kw = (16 + kb) * 32 + hf * 16;
            V16 vz, vr, vq;
            vz.q[0] = *(const uint4*)(wz + kw); vz.q[1] = *(const uint4*)(wz + kw + 8);
            vr.q[0] = *(const uint4*)(wr + kw); vr.q[1] = *(const uint4*)(wr + kw + 8);
            vq.q[0] = *(const uint4*)(wq + kw); vq.q[1] = *(const uint4*)(wq + kw + 8);

            az = __builtin_amdgcn_wmma_f32_16x16x32_bf16(false, a.bf, false, vz.bf, (short)0, az, false, false);
            ar = __builtin_amdgcn_wmma_f32_16x16x32_bf16(false, a.bf, false, vr.bf, (short)0, ar, false, false);
            ac = __builtin_amdgcn_wmma_f32_16x16x32_bf16(false, a.bf, false, vq.bf, (short)0, ac, false, false);
        }

        // ---- epilogue: gates -> h_t (fp32 carry in registers) ----
        #pragma unroll
        for (int v = 0; v < 8; ++v) {
            const int b = bbase + v;
            float z = sigmoidf_(az[v]);
            float r = sigmoidf_(ar[v]);
            float c = tanhf_(ac[v]);
            float ht = z * hreg[v] + (1.0f - z) * (r * c);
            hreg[v] = ht;
            y[((size_t)b * T_ + t) * H_ + j] = ht;
            hwr[(size_t)b * H_ + j] = f2bf(ht);
        }

        // ---- device-wide barrier (generation counter) ----
        __threadfence();
        __syncthreads();
        if (threadIdx.x == 0) {
            __hip_atomic_fetch_add(cnt, 1u, __ATOMIC_RELEASE, __HIP_MEMORY_SCOPE_AGENT);
            const unsigned target = (unsigned)(t + 1) * (unsigned)gridDim.x;
            while (__hip_atomic_load(cnt, __ATOMIC_ACQUIRE, __HIP_MEMORY_SCOPE_AGENT) < target) {
                __builtin_amdgcn_s_sleep(1);
            }
        }
        __syncthreads();
    }

    // final hidden state from registers
    #pragma unroll
    for (int v = 0; v < 8; ++v) hlast[(size_t)(bbase + v) * H_ + j] = hreg[v];
}

extern "C" void kernel_launch(void* const* d_in, const int* in_sizes, int n_in,
                              void* d_out, int out_size, void* d_ws, size_t ws_size,
                              hipStream_t stream) {
    const float* x  = (const float*)d_in[0];
    const float* h0 = (const float*)d_in[1];
    const float* Wi = (const float*)d_in[2];
    const float* bi = (const float*)d_in[3];
    const float* Wh = (const float*)d_in[4];
    const float* bh = (const float*)d_in[5];

    float* y     = (float*)d_out;                       // [B,T,H]
    float* hlast = y + (size_t)B_ * T_ * H_;            // [B,H]

    char* ws = (char*)d_ws;
    unsigned short* Wc   = (unsigned short*)(ws + WC_OFF);
    float*          bcw  = (float*)(ws + BC_OFF);
    unsigned short* hb0  = (unsigned short*)(ws + HB0_OFF);
    unsigned short* hb1  = (unsigned short*)(ws + HB1_OFF);
    unsigned*       cnt  = (unsigned*)(ws + CNT_OFF);
    unsigned short* xbf  = (unsigned short*)(ws + XBF_OFF);

    const int use_xbf = (ws_size >= WS_NEED_XBF) ? 1 : 0;

    {
        int n = G3 * KC;
        gru_pack_kernel<<<(n + 255) / 256, 256, 0, stream>>>(Wi, Wh, bi, bh, Wc, bcw);
    }
    if (use_xbf) {
        size_t n8 = (size_t)B_ * T_ * I_ / 8;
        gru_xpack_kernel<<<(unsigned)((n8 + 255) / 256), 256, 0, stream>>>(x, xbf);
    }
    {
        int n = B_ * H_;
        gru_init_kernel<<<(n + 255) / 256, 256, 0, stream>>>(h0, hb0, cnt);
    }
    gru_scan_kernel<<<NBLOCKS, NTHREADS, 0, stream>>>(
        x, xbf, use_xbf, Wc, bcw, h0, hb0, hb1, y, hlast, cnt);
}
